// HyenaBaseBlock_62139586839023
// MI455X (gfx1250) — matrix operand order repaired and checked by reading, not compiled, not run on hardware
//
#include <hip/hip_runtime.h>

// ---------------- problem dims ----------------
constexpr int kB  = 8;
constexpr int kL  = 4096;
constexpr int kD  = 1024;
constexpr int kDP = 128;
constexpr int kFF = 2048;
constexpr int kN  = 8192;     // FFT size = 2*L
constexpr int kLog2N = 13;
constexpr float kEPS = 1e-5f;

typedef __bf16 bf16_t;
typedef __attribute__((ext_vector_type(16))) __bf16 v16bf;
typedef __attribute__((ext_vector_type(8)))  float  v8f;

struct cplx { float re, im; };

#define WMMA_BF16(a, b, c) \
  __builtin_amdgcn_wmma_f32_16x16x32_bf16(false, (a), false, (b), (short)0, (c), false, false)

// ---------------- reductions ----------------
__device__ __forceinline__ float blk_reduce_sum(float v, float* sbuf) {
#pragma unroll
  for (int off = 16; off > 0; off >>= 1) v += __shfl_xor(v, off, 32);
  int lane = threadIdx.x & 31, w = threadIdx.x >> 5;
  if (lane == 0) sbuf[w] = v;
  __syncthreads();
  float r = 0.0f;
  if (w == 0) {
    r = (lane < (int)(blockDim.x >> 5)) ? sbuf[lane] : 0.0f;
#pragma unroll
    for (int off = 16; off > 0; off >>= 1) r += __shfl_xor(r, off, 32);
    if (lane == 0) sbuf[0] = r;
  }
  __syncthreads();
  r = sbuf[0];
  __syncthreads();
  return r;
}

// ---------------- weight conversion ----------------
__global__ void cvt_bf16_kernel(const float* __restrict__ in, bf16_t* __restrict__ out, int n) {
  int i = blockIdx.x * blockDim.x + threadIdx.x;
  if (i < n) out[i] = (bf16_t)in[i];
}

// ---------------- LayerNorm over last dim (row = one (b,l)) ----------------
__global__ __launch_bounds__(256) void ln_rows_kernel(
    const float* __restrict__ in, const float* __restrict__ gamma,
    const float* __restrict__ beta, float* __restrict__ out) {
  __shared__ float sbuf[32];
  size_t row = blockIdx.x;
  const float* r = in + row * kD;
  float lsum = 0.f, lsq = 0.f;
  for (int i = threadIdx.x; i < kD; i += blockDim.x) { float v = r[i]; lsum += v; lsq += v * v; }
  float s  = blk_reduce_sum(lsum, sbuf);
  float sq = blk_reduce_sum(lsq, sbuf);
  float mu = s * (1.0f / kD);
  float rstd = rsqrtf(sq * (1.0f / kD) - mu * mu + kEPS);
  float* o = out + row * kD;
  for (int i = threadIdx.x; i < kD; i += blockDim.x)
    o[i] = (r[i] - mu) * rstd * gamma[i] + beta[i];
}

// ---------------- LN + causal decay window (filter rows over L) ----------------
__global__ __launch_bounds__(256) void ln_window_kernel(
    const float* __restrict__ g, const float* __restrict__ gamma,
    const float* __restrict__ beta, const float* __restrict__ a,
    float* __restrict__ filt) {
  __shared__ float sbuf[32];
  size_t row = blockIdx.x;
  const float* r = g + row * kD;
  float lsum = 0.f, lsq = 0.f;
  for (int i = threadIdx.x; i < kD; i += blockDim.x) { float v = r[i]; lsum += v; lsq += v * v; }
  float s  = blk_reduce_sum(lsum, sbuf);
  float sq = blk_reduce_sum(lsq, sbuf);
  float mu = s * (1.0f / kD);
  float rstd = rsqrtf(sq * (1.0f / kD) - mu * mu + kEPS);
  float win = expf(-(float)row * expf(a[0]));
  float* o = filt + row * kD;
  for (int i = threadIdx.x; i < kD; i += blockDim.x)
    o[i] = ((r[i] - mu) * rstd * gamma[i] + beta[i]) * win;
}

// ---------------- WMMA fragment conventions ----------------
// A (16x32 bf16, M x K): m = lane&15 ; K base = 32*kc + 8*(lane>>4), +16 for 2nd half
// B (32x16 bf16, K x N) for X @ W^T with row-major W(N,K):
//   lane supplies W[ntile*16 + (lane&15)][32*kc + 8*(lane>>4) + 0..7, +16..23] (same as A)
// C/D (16x16 f32): element r of v8f -> M = r + 8*(lane>>4), N = lane&15

// ---------------- pos @ w_pos^T (L x DP) x (D x DP)^T ----------------
__global__ __launch_bounds__(256) void posmm_kernel(
    const float* __restrict__ pos, const bf16_t* __restrict__ wposb,
    float* __restrict__ g) {
  __shared__ bf16_t sA[16][kDP];          // 4 KB
  int row0 = blockIdx.x * 16;
  for (int idx = threadIdx.x; idx < 16 * kDP; idx += blockDim.x) {
    int rr = idx >> 7, k = idx & (kDP - 1);
    sA[rr][k] = (bf16_t)pos[(size_t)(row0 + rr) * kDP + k];
  }
  __syncthreads();

  int lane = threadIdx.x & 31, w = threadIdx.x >> 5;
  int m = lane & 15, kgrp = lane >> 4;
  for (int t = 0; t < 8; ++t) {                  // 64 n-tiles / 8 waves
    int ntile = w * 8 + t;
    int n = ntile * 16 + m;
    v8f acc = {};
    for (int kc = 0; kc < kDP / 32; ++kc) {
      v16bf afrag, bfrag;
      int ka = kc * 32 + kgrp * 8;
#pragma unroll
      for (int i = 0; i < 8; ++i) { afrag[i] = sA[m][ka + i]; afrag[8 + i] = sA[m][ka + 16 + i]; }
      const bf16_t* bp = wposb + (size_t)n * kDP + kc * 32 + kgrp * 8;
#pragma unroll
      for (int i = 0; i < 16; ++i) bfrag[i] = bp[(i < 8) ? i : i + 8];
      acc = WMMA_BF16(afrag, bfrag, acc);
    }
    int cm = row0 + kgrp * 8;
#pragma unroll
    for (int r = 0; r < 8; ++r) g[(size_t)(cm + r) * kD + n] = acc[r];
  }
}

// ---------------- in-LDS 8192-pt FFT (DIF fwd, bit-reversed out) ----------------
__device__ __forceinline__ void fft8192_dif(cplx* s) {
  for (int stage = kLog2N; stage >= 1; --stage) {
    int mh = 1 << (stage - 1);
    for (int bf = threadIdx.x; bf < kN / 2; bf += blockDim.x) {
      int j = bf & (mh - 1);
      int i0 = ((bf >> (stage - 1)) << stage) + j;
      int i1 = i0 + mh;
      cplx u = s[i0], v = s[i1];
      float ang = -6.28318530718f * (float)j / (float)(mh << 1);
      float c = __cosf(ang), sn = __sinf(ang);
      float dr = u.re - v.re, di = u.im - v.im;
      s[i0].re = u.re + v.re; s[i0].im = u.im + v.im;
      s[i1].re = dr * c - di * sn;
      s[i1].im = dr * sn + di * c;
    }
    __syncthreads();
  }
}

// inverse DIT: bit-reversed input -> natural order output (no 1/N scale here)
__device__ __forceinline__ void fft8192_inv_dit(cplx* s) {
  for (int stage = 1; stage <= kLog2N; ++stage) {
    int mh = 1 << (stage - 1);
    for (int bf = threadIdx.x; bf < kN / 2; bf += blockDim.x) {
      int j = bf & (mh - 1);
      int i0 = ((bf >> (stage - 1)) << stage) + j;
      int i1 = i0 + mh;
      float ang = 6.28318530718f * (float)j / (float)(mh << 1);
      float c = __cosf(ang), sn = __sinf(ang);
      cplx t = s[i1], u = s[i0];
      float tr = t.re * c - t.im * sn;
      float ti = t.re * sn + t.im * c;
      s[i0].re = u.re + tr; s[i0].im = u.im + ti;
      s[i1].re = u.re - tr; s[i1].im = u.im - ti;
    }
    __syncthreads();
  }
}

// ---------------- filter spectrum per channel (stored bit-reversed) ----------------
__global__ __launch_bounds__(256) void fft_filter_kernel(
    const float* __restrict__ filt, cplx* __restrict__ wfh) {
  __shared__ cplx s[kN];                  // 64 KB
  int d = blockIdx.x;
  for (int l = threadIdx.x; l < kN; l += blockDim.x) {
    float re = (l < kL) ? filt[(size_t)l * kD + d] : 0.0f;
    s[l].re = re; s[l].im = 0.0f;
  }
  __syncthreads();
  fft8192_dif(s);
  cplx* o = wfh + (size_t)d * kN;
  for (int k = threadIdx.x; k < kN; k += blockDim.x) o[k] = s[k];
}

// ---------------- per (b,d) FFT convolution, fully LDS-resident ----------------
__global__ __launch_bounds__(256) void fft_conv_kernel(
    const float* __restrict__ zn, const cplx* __restrict__ wfh,
    float* __restrict__ cwhz) {
  __shared__ cplx s[kN];                  // 64 KB
  int bd = blockIdx.x;
  int b = bd >> 10;            // / kD
  int d = bd & (kD - 1);
  const float* zc = zn + (size_t)b * kL * kD + d;
  for (int l = threadIdx.x; l < kN; l += blockDim.x) {
    float re = (l < kL) ? zc[(size_t)l * kD] : 0.0f;
    s[l].re = re; s[l].im = 0.0f;
  }
  __syncthreads();
  fft8192_dif(s);
  const cplx* h = wfh + (size_t)d * kN;   // same bit-reversed order -> multiply directly
  for (int k = threadIdx.x; k < kN; k += blockDim.x) {
    cplx a = s[k], w = h[k];
    s[k].re = a.re * w.re - a.im * w.im;
    s[k].im = a.re * w.im + a.im * w.re;
  }
  __syncthreads();
  fft8192_inv_dit(s);
  float* oc = cwhz + (size_t)b * kL * kD + d;
  const float inv = 1.0f / (float)kN;
  for (int l = threadIdx.x; l < kL; l += blockDim.x)
    oc[(size_t)l * kD] = s[l].re * inv;
}

// ---------------- per-wave LN of rows into LDS (bf16) ----------------
__device__ __forceinline__ void ln_rows_to_lds(
    const float* __restrict__ src, int row0, int nrows,
    const float* __restrict__ gamma, const float* __restrict__ beta,
    bf16_t (*sX)[kD]) {
  int lane = threadIdx.x & 31, w = threadIdx.x >> 5;
  for (int rr = w; rr < nrows; rr += 8) {
    const float* xr = src + (size_t)(row0 + rr) * kD;
    float lsum = 0.f, lsq = 0.f;
    for (int i = lane; i < kD; i += 32) { float v = xr[i]; lsum += v; lsq += v * v; }
#pragma unroll
    for (int off = 16; off > 0; off >>= 1) {
      lsum += __shfl_xor(lsum, off, 32); lsq += __shfl_xor(lsq, off, 32);
    }
    float mu = lsum * (1.0f / kD);
    float rstd = rsqrtf(lsq * (1.0f / kD) - mu * mu + kEPS);
    for (int i = lane; i < kD; i += 32)
      sX[rr][i] = (bf16_t)((xr[i] - mu) * rstd * gamma[i] + beta[i]);
  }
}

// ---------------- gate: LN(x) @ w_mx^T, y = xg*cwhz + z ----------------
// 32 rows / block; K-outer register-blocked: each B fragment feeds 2 m-tiles.
__global__ __launch_bounds__(256) void gate_kernel(
    const float* __restrict__ x, const float* __restrict__ gamma,
    const float* __restrict__ beta, const bf16_t* __restrict__ wmx,
    const float* __restrict__ cwhz, const float* __restrict__ z,
    float* __restrict__ y) {
  __shared__ bf16_t sX[32][kD];           // 64 KB
  int row0 = blockIdx.x * 32;
  ln_rows_to_lds(x, row0, 32, gamma, beta, sX);
  __syncthreads();

  int lane = threadIdx.x & 31, w = threadIdx.x >> 5;
  int m = lane & 15, kgrp = lane >> 4;

  v8f zero = {};
  v8f acc[8][2];
#pragma unroll
  for (int t = 0; t < 8; ++t) { acc[t][0] = zero; acc[t][1] = zero; }

  for (int kc = 0; kc < kD / 32; ++kc) {
    int ka = kc * 32 + kgrp * 8;
    v16bf a0, a1;
#pragma unroll
    for (int i = 0; i < 8; ++i) {
      a0[i] = sX[m][ka + i];       a0[8 + i] = sX[m][ka + 16 + i];
      a1[i] = sX[16 + m][ka + i];  a1[8 + i] = sX[16 + m][ka + 16 + i];
    }
#pragma unroll
    for (int t = 0; t < 8; ++t) {
      int n = (w * 8 + t) * 16 + m;
      const bf16_t* bp = wmx + (size_t)n * kD + kc * 32 + kgrp * 8;
      __builtin_prefetch(bp + 32, 0, 1);         // global_prefetch next K-chunk
      v16bf bfrag;
#pragma unroll
      for (int i = 0; i < 16; ++i) bfrag[i] = bp[(i < 8) ? i : i + 8];
      acc[t][0] = WMMA_BF16(a0, bfrag, acc[t][0]);
      acc[t][1] = WMMA_BF16(a1, bfrag, acc[t][1]);
    }
  }

#pragma unroll
  for (int t = 0; t < 8; ++t) {
    int n = (w * 8 + t) * 16 + m;
#pragma unroll
    for (int mt = 0; mt < 2; ++mt) {
      int cm = row0 + mt * 16 + kgrp * 8;
#pragma unroll
      for (int r = 0; r < 8; ++r) {
        size_t idx = (size_t)(cm + r) * kD + n;
        y[idx] = acc[t][mt][r] * cwhz[idx] + z[idx];
      }
    }
  }
}

// ---------------- FFN stage 1: silu(LN(y) @ w1^T + b1) -> bf16 ----------------
// 32 rows / block; 128 n-tiles done as 2 passes of (8 waves x 8 tiles).
__global__ __launch_bounds__(256) void ffn1_kernel(
    const float* __restrict__ y, const float* __restrict__ gamma,
    const float* __restrict__ beta, const bf16_t* __restrict__ w1,
    const float* __restrict__ b1, bf16_t* __restrict__ hid) {
  __shared__ bf16_t sX[32][kD];           // 64 KB
  int row0 = blockIdx.x * 32;
  ln_rows_to_lds(y, row0, 32, gamma, beta, sX);
  __syncthreads();

  int lane = threadIdx.x & 31, w = threadIdx.x >> 5;
  int m = lane & 15, kgrp = lane >> 4;
  v8f zero = {};

  for (int pass = 0; pass < 2; ++pass) {
    v8f acc[8][2];
#pragma unroll
    for (int t = 0; t < 8; ++t) { acc[t][0] = zero; acc[t][1] = zero; }

    for (int kc = 0; kc < kD / 32; ++kc) {
      int ka = kc * 32 + kgrp * 8;
      v16bf a0, a1;
#pragma unroll
      for (int i = 0; i < 8; ++i) {
        a0[i] = sX[m][ka + i];       a0[8 + i] = sX[m][ka + 16 + i];
        a1[i] = sX[16 + m][ka + i];  a1[8 + i] = sX[16 + m][ka + 16 + i];
      }
#pragma unroll
      for (int t = 0; t < 8; ++t) {
        int n = (pass * 64 + w * 8 + t) * 16 + m;
        const bf16_t* bp = w1 + (size_t)n * kD + kc * 32 + kgrp * 8;
        v16bf bfrag;
#pragma unroll
        for (int i = 0; i < 16; ++i) bfrag[i] = bp[(i < 8) ? i : i + 8];
        acc[t][0] = WMMA_BF16(a0, bfrag, acc[t][0]);
        acc[t][1] = WMMA_BF16(a1, bfrag, acc[t][1]);
      }
    }

#pragma unroll
    for (int t = 0; t < 8; ++t) {
      int n = (pass * 64 + w * 8 + t) * 16 + m;
      float bias = b1[n];
#pragma unroll
      for (int mt = 0; mt < 2; ++mt) {
        int cm = row0 + mt * 16 + kgrp * 8;
#pragma unroll
        for (int r = 0; r < 8; ++r) {
          float v = acc[t][mt][r] + bias;
          float sv = v / (1.0f + expf(-v));      // silu
          hid[(size_t)(cm + r) * kFF + n] = (bf16_t)sv;
        }
      }
    }
  }
}

// ---------------- FFN stage 2: hid @ w2^T + b2 + y -> out ----------------
// 32 rows / block; A (already bf16 in ws) read directly from global,
// loaded once per kc and reused across all 8 n-tiles (K-outer, acc-resident).
__global__ __launch_bounds__(256) void ffn2_kernel(
    const bf16_t* __restrict__ hid, const bf16_t* __restrict__ w2,
    const float* __restrict__ b2, const float* __restrict__ y,
    float* __restrict__ out) {
  int row0 = blockIdx.x * 32;
  int lane = threadIdx.x & 31, w = threadIdx.x >> 5;
  int m = lane & 15, kgrp = lane >> 4;

  v8f zero = {};
  v8f acc[8][2];
#pragma unroll
  for (int t = 0; t < 8; ++t) { acc[t][0] = zero; acc[t][1] = zero; }

  const bf16_t* arow0 = hid + (size_t)(row0 + m) * kFF;
  const bf16_t* arow1 = hid + (size_t)(row0 + 16 + m) * kFF;

  for (int kc = 0; kc < kFF / 32; ++kc) {
    int ka = kc * 32 + kgrp * 8;
    v16bf a0, a1;
#pragma unroll
    for (int i = 0; i < 8; ++i) {
      a0[i] = arow0[ka + i];  a0[8 + i] = arow0[ka + 16 + i];
      a1[i] = arow1[ka + i];  a1[8 + i] = arow1[ka + 16 + i];
    }
#pragma unroll
    for (int t = 0; t < 8; ++t) {
      int n = (w * 8 + t) * 16 + m;
      const bf16_t* bp = w2 + (size_t)n * kFF + kc * 32 + kgrp * 8;
      __builtin_prefetch(bp + 32, 0, 1);
      v16bf bfrag;
#pragma unroll
      for (int i = 0; i < 16; ++i) bfrag[i] = bp[(i < 8) ? i : i + 8];
      acc[t][0] = WMMA_BF16(a0, bfrag, acc[t][0]);
      acc[t][1] = WMMA_BF16(a1, bfrag, acc[t][1]);
    }
  }

#pragma unroll
  for (int t = 0; t < 8; ++t) {
    int n = (w * 8 + t) * 16 + m;
    float bias = b2[n];
#pragma unroll
    for (int mt = 0; mt < 2; ++mt) {
      int cm = row0 + mt * 16 + kgrp * 8;
#pragma unroll
      for (int r = 0; r < 8; ++r) {
        size_t idx = (size_t)(cm + r) * kD + n;
        out[idx] = acc[t][mt][r] + bias + y[idx];
      }
    }
  }
}

// ---------------- host launcher ----------------
extern "C" void kernel_launch(void* const* d_in, const int* in_sizes, int n_in,
                              void* d_out, int out_size, void* d_ws, size_t ws_size,
                              hipStream_t stream) {
  const float* z     = (const float*)d_in[0];
  const float* x     = (const float*)d_in[1];
  const float* pos   = (const float*)d_in[2];
  const float* a     = (const float*)d_in[3];
  const float* wpos  = (const float*)d_in[4];
  const float* gamma = (const float*)d_in[5];
  const float* beta  = (const float*)d_in[6];
  const float* wmx   = (const float*)d_in[7];
  const float* w1    = (const float*)d_in[8];
  const float* b1    = (const float*)d_in[9];
  const float* w2    = (const float*)d_in[10];
  const float* b2    = (const float*)d_in[11];
  float* out = (float*)d_out;

  const size_t BLD = (size_t)kB * kL * kD;
  char* p = (char*)d_ws;
  auto alloc = [&](size_t bytes) -> char* {
    char* r = p; p += (bytes + 255) & ~(size_t)255; return r;
  };
  float*  zn    = (float*) alloc(BLD * sizeof(float));
  float*  yb    = (float*) alloc(BLD * sizeof(float));
  float*  cwhz  = (float*) alloc(BLD * sizeof(float));
  float*  gbuf  = (float*) alloc((size_t)kL * kD * sizeof(float));
  float*  filt  = (float*) alloc((size_t)kL * kD * sizeof(float));
  cplx*   wfh   = (cplx*)  alloc((size_t)kD * kN * sizeof(cplx));
  bf16_t* hid   = (bf16_t*)alloc((size_t)kB * kL * kFF * sizeof(bf16_t));
  bf16_t* wmxb  = (bf16_t*)alloc((size_t)kD * kD * sizeof(bf16_t));
  bf16_t* w1b   = (bf16_t*)alloc((size_t)kFF * kD * sizeof(bf16_t));
  bf16_t* w2b   = (bf16_t*)alloc((size_t)kD * kFF * sizeof(bf16_t));
  bf16_t* wposb = (bf16_t*)alloc((size_t)kD * kDP * sizeof(bf16_t));

  // weight precision conversion (tiny)
  cvt_bf16_kernel<<<(kD * kD + 255) / 256, 256, 0, stream>>>(wmx, wmxb, kD * kD);
  cvt_bf16_kernel<<<(kFF * kD + 255) / 256, 256, 0, stream>>>(w1, w1b, kFF * kD);
  cvt_bf16_kernel<<<(kD * kFF + 255) / 256, 256, 0, stream>>>(w2, w2b, kD * kFF);
  cvt_bf16_kernel<<<(kD * kDP + 255) / 256, 256, 0, stream>>>(wpos, wposb, kD * kDP);

  // zn = LN(z)
  ln_rows_kernel<<<kB * kL, 256, 0, stream>>>(z, gamma, beta, zn);
  // implicit filter: g = pos @ w_pos^T ; filt = LN(g) * window
  posmm_kernel<<<kL / 16, 256, 0, stream>>>(pos, wposb, gbuf);
  ln_window_kernel<<<kL, 256, 0, stream>>>(gbuf, gamma, beta, a, filt);
  // filter spectrum per channel (bit-reversed order)
  fft_filter_kernel<<<kD, 256, 0, stream>>>(filt, wfh);
  // FFT convolution per (b, d)
  fft_conv_kernel<<<kB * kD, 256, 0, stream>>>(zn, wfh, cwhz);
  // gate + residual
  gate_kernel<<<kB * kL / 32, 256, 0, stream>>>(x, gamma, beta, wmxb, cwhz, z, yb);
  // FFN
  ffn1_kernel<<<kB * kL / 32, 256, 0, stream>>>(yb, gamma, beta, w1b, b1, hid);
  ffn2_kernel<<<kB * kL / 32, 256, 0, stream>>>(hid, w2b, b2, yb, out);
}
